// myGCNmodule_28106265985138
// MI455X (gfx1250) — compile-verified, matrix operand order checked by reading, not run on hardware
//
#include <hip/hip_runtime.h>

// ---------------- problem constants ----------------
#define NN   10000
#define DD   15
#define EE   320
#define OUTD 30
#define HH   5
#define HDIM 64
#define GPW  2          // nodes per workgroup
#define KT   10         // K tiles of 32 (320/32)
#define NT   20         // N tiles of 16 (320/16)
#define MSZ  (EE*EE)    // elements per swizzled weight matrix (102400)
#define SMEM_BYTES (6*10240*2 + 2560*4)   // 133120 B dynamic LDS

#define AS1 __attribute__((address_space(1)))   // force GLOBAL (not FLAT) vmem path

typedef __attribute__((ext_vector_type(16))) __bf16 v16bf;
typedef __attribute__((ext_vector_type(8)))  float  v8f;
typedef __attribute__((ext_vector_type(4)))  unsigned u32x4;   // POD 128-bit chunk

union BFrag { v16bf v; unsigned u[8]; unsigned short s[16]; };

__device__ __forceinline__ unsigned short f2bf(float f) {
  union { float f; unsigned u; } c; c.f = f;
  unsigned u = c.u;
  unsigned r = u + 0x7FFFu + ((u >> 16) & 1u);   // round-to-nearest-even
  return (unsigned short)(r >> 16);
}
__device__ __forceinline__ float bf2f(unsigned short h) {
  union { unsigned u; float f; } c; c.u = ((unsigned)h) << 16; return c.f;
}

// Invert the WMMA 16-bit B-matrix VGPR layout: element (k,n) of a 32x16 tile
// lives at lane = (n&15) + 16*half, vector slot 2p+low  (ISA 7.12.2).
__device__ __forceinline__ int sw_idx(int k, int n, int ntiles) {
  int kt = k >> 5, kk = k & 31;
  int nt = n >> 4, nn = n & 15;
  int half = (kk & 15) >> 3;
  int p    = ((kk & 7) >> 1) + ((kk >= 16) ? 4 : 0);
  int low  = kk & 1;
  int lane = nn + 16 * half;
  return ((kt * ntiles + nt) * 32 + lane) * 16 + 2 * p + low;
}

// ---------------- prep kernels ----------------
__global__ void k_xbf(const float* __restrict__ X, unsigned short* __restrict__ Xbf, int n) {
  int i = blockIdx.x * 256 + threadIdx.x;
  if (i < n) Xbf[i] = f2bf(X[i]);
}

// combined C = Wm @ wm.T  (fuses custom proj with MHA qkv proj), stored swizzled
__global__ void k_combined(const float* in_Wq, const float* in_Wk, const float* in_Wv,
                           const float* in_qkv_w,
                           const float* out_Wq, const float* out_Wk, const float* out_Wv,
                           const float* out_qkv_w,
                           unsigned short* __restrict__ SW) {
  int m  = blockIdx.y;                       // 0..5
  int id = blockIdx.x * 256 + threadIdx.x;   // 0..E*E-1
  if (id >= EE * EE) return;
  int k = id / EE, n = id % EE;
  const float* Wm = in_Wq; const float* qkv = in_qkv_w; int off = 0;
  switch (m) {
    case 0: Wm = in_Wq;  qkv = in_qkv_w;  off = 0;      break;
    case 1: Wm = in_Wk;  qkv = in_qkv_w;  off = EE;     break;
    case 2: Wm = in_Wv;  qkv = in_qkv_w;  off = 2 * EE; break;
    case 3: Wm = out_Wq; qkv = out_qkv_w; off = 0;      break;
    case 4: Wm = out_Wk; qkv = out_qkv_w; off = EE;     break;
    default: Wm = out_Wv; qkv = out_qkv_w; off = 2 * EE; break;
  }
  const AS1 float* wrow = (const AS1 float*)(Wm + k * EE);
  const AS1 float* qrow = (const AS1 float*)(qkv + (off + n) * EE);
  float acc = 0.f;
  #pragma unroll 8
  for (int j = 0; j < EE; ++j) acc += wrow[j] * qrow[j];
  int dst = (m < 3) ? m : m + 1;             // slots 0,1,2, 4,5,6
  SW[dst * MSZ + sw_idx(k, n, NT)] = f2bf(acc);
}

// transposed weights (used as X @ w.T), stored swizzled
__global__ void k_transpose_sw(const float* in_o_w, const float* out_o_w,
                               const float* fin_qkv_w,
                               unsigned short* __restrict__ SW) {
  int m  = blockIdx.y;                       // 0..3
  int id = blockIdx.x * 256 + threadIdx.x;
  if (id >= EE * EE) return;
  int k = id / EE, n = id % EE;
  float v; int dst;
  switch (m) {
    case 0:  v = in_o_w[n * EE + k];                 dst = 3; break;
    case 1:  v = out_o_w[n * EE + k];                dst = 7; break;
    case 2:  v = fin_qkv_w[(EE + n) * EE + k];       dst = 8; break;   // wk
    default: v = fin_qkv_w[(2 * EE + n) * EE + k];   dst = 9; break;   // wv
  }
  SW[dst * MSZ + sw_idx(k, n, NT)] = f2bf(v);
}

// ---------------- WMMA GEMM: D[M x 320] = A[M x 320] @ Bsw + bias ----------------
// A: bf16 rows in LDS (row ptr via arow_of).  B: swizzled bf16 in global (AS1).
// MT = M/16 (2 or 4). 8 waves split (M-tile, 5-wide N-tile groups).
template <int MT, typename ARow>
__device__ __forceinline__ void gemm_sw(ARow arow_of, const unsigned short* __restrict__ Bsw,
                                        const float* __restrict__ bias,
                                        unsigned short* __restrict__ D, int ldD, int tid) {
  const int lane = tid & 31, wave = tid >> 5;
  const int mt   = wave % MT;
  const int grp  = wave / MT;
  constexpr int GROUPS = 8 / MT;
  constexpr int NTP    = NT / GROUPS;        // 5 (MT=2) or 10 (MT=4)
  const int n15  = lane & 15;
  const int half = lane >> 4;
  const unsigned short* arow = arow_of(mt * 16 + n15);
  const AS1 unsigned short* Bg = (const AS1 unsigned short*)Bsw;   // GLOBAL path
  const AS1 float* biasg = (const AS1 float*)bias;

  for (int pass = 0; pass < NTP; pass += 5) {
    const int ntBase = grp * NTP + pass;
    v8f acc[5];
    #pragma unroll
    for (int j = 0; j < 5; ++j) acc[j] = (v8f){0.f,0.f,0.f,0.f,0.f,0.f,0.f,0.f};

    #pragma unroll 2
    for (int kt = 0; kt < KT; ++kt) {
      BFrag af;
      #pragma unroll
      for (int p = 0; p < 8; ++p) {          // ISA 16-bit A layout (7.12.2)
        int k = (p < 4) ? (2 * p + 8 * half) : (2 * p + 8 + 8 * half);
        af.u[p] = *(const unsigned*)(arow + kt * 32 + k);
      }
      if (kt + 1 < KT)                       // warm next weight K-tile in L2/L0
        __builtin_prefetch(Bsw + (((kt + 1) * NT + ntBase) * 32) * 16, 0, 1);
      #pragma unroll
      for (int j = 0; j < 5; ++j) {
        BFrag bf;
        bf.v = *(const AS1 v16bf*)(Bg + (((kt * NT + ntBase + j) * 32 + lane) * 16));
        acc[j] = __builtin_amdgcn_wmma_f32_16x16x32_bf16(
            false, af.v, false, bf.v, (short)0, acc[j], false, false);
      }
    }
    #pragma unroll
    for (int j = 0; j < 5; ++j) {
      int col = (ntBase + j) * 16 + n15;
      float b = bias ? biasg[col] : 0.f;
      #pragma unroll
      for (int r = 0; r < 8; ++r) {          // C/D layout: M = r + 8*half, N = lane&15
        int row = mt * 16 + r + 8 * half;
        D[row * ldD + col] = f2bf(acc[j][r] + b);
      }
    }
  }
}

// ---------------- fused main kernel: 2 nodes per workgroup ----------------
__global__ void __launch_bounds__(256)
k_main(const unsigned short* __restrict__ Xbf, const unsigned short* __restrict__ SW,
       const int* __restrict__ in_idx, const int* __restrict__ out_idx,
       const float* __restrict__ in_qkv_b, const float* __restrict__ in_o_b,
       const float* __restrict__ out_qkv_b, const float* __restrict__ out_o_b,
       const float* __restrict__ fin_qkv_w, const float* __restrict__ fin_qkv_b,
       const float* __restrict__ fin_o_w, const float* __restrict__ fin_o_b,
       const float* __restrict__ Wfin, float* __restrict__ out) {
  extern __shared__ char smem[];
  unsigned short* sSeq  = (unsigned short*)smem;     // 32x320
  unsigned short* sQ    = sSeq + 10240;
  unsigned short* sK    = sQ + 10240;
  unsigned short* sV    = sK + 10240;
  unsigned short* sAin  = sV + 10240;
  unsigned short* sAout = sAin + 10240;
  float* sScr = (float*)(sAout + 10240);             // 2560 f32
  unsigned short* sK2f = sSeq;                       // phase-3 alias: 64x320
  unsigned short* sV2f = sK;                         // phase-3 alias: 64x320

  const int tid   = threadIdx.x;
  const int node0 = blockIdx.x * GPW;

  const int*   idxs[2] = { in_idx, out_idx };
  const float* qkvb[2] = { in_qkv_b, out_qkv_b };
  const float* ob[2]   = { in_o_b, out_o_b };
  const unsigned short* Cq[2] = { SW + 0 * MSZ, SW + 4 * MSZ };
  const unsigned short* Ck[2] = { SW + 1 * MSZ, SW + 5 * MSZ };
  const unsigned short* Cv[2] = { SW + 2 * MSZ, SW + 6 * MSZ };
  const unsigned short* Ow[2] = { SW + 3 * MSZ, SW + 7 * MSZ };
  unsigned short* aDst[2] = { sAin, sAout };

  auto rowSeq = [&](int r) -> const unsigned short* { return sSeq + r * EE; };

  for (int L = 0; L < 2; ++L) {
    // gather [self ; neighbors] -> 32 bf16 rows in LDS (128-bit copies)
    {
      int r = tid >> 3, ch = tid & 7;
      int g = r >> 4, l = r & 15;
      int node = node0 + g;
      int src = (l == 0) ? node : ((const AS1 int*)idxs[L])[node * DD + (l - 1)];
      const AS1 u32x4* sp = (const AS1 u32x4*)(Xbf + (size_t)src * EE) + ch * 5;
      u32x4* dp = (u32x4*)(sSeq + r * EE) + ch * 5;
      #pragma unroll
      for (int q = 0; q < 5; ++q) dp[q] = sp[q];
    }
    __syncthreads();

    // fused projections q2/k2/v2 = seq @ (W? . w?.T) + b?
    gemm_sw<2>(rowSeq, Cq[L], qkvb[L],          sQ, EE, tid);
    gemm_sw<2>(rowSeq, Ck[L], qkvb[L] + EE,     sK, EE, tid);
    gemm_sw<2>(rowSeq, Cv[L], qkvb[L] + 2 * EE, sV, EE, tid);
    __syncthreads();

    // scores (5 heads, hd=64, scale 1/8) -> f32 LDS
    for (int i = tid; i < GPW * HH * 16 * 16; i += 256) {
      int m = i & 15, l = (i >> 4) & 15, t = i >> 8, h = t % HH, g = t / HH;
      const unsigned short* qr = sQ + (g * 16 + l) * EE + h * HDIM;
      const unsigned short* kr = sK + (g * 16 + m) * EE + h * HDIM;
      float acc = 0.f;
      #pragma unroll 8
      for (int k = 0; k < HDIM; ++k) acc += bf2f(qr[k]) * bf2f(kr[k]);
      sScr[i] = acc * 0.125f;
    }
    __syncthreads();
    if (tid < GPW * HH * 16) {                 // softmax per row
      float* row = sScr + tid * 16;
      float mx = row[0];
      #pragma unroll
      for (int m = 1; m < 16; ++m) mx = fmaxf(mx, row[m]);
      float s = 0.f;
      #pragma unroll
      for (int m = 0; m < 16; ++m) { float e = __expf(row[m] - mx); row[m] = e; s += e; }
      float inv = 1.f / s;
      #pragma unroll
      for (int m = 0; m < 16; ++m) row[m] *= inv;
    }
    __syncthreads();
    // AV -> o (reuse sSeq)
    for (int i = tid; i < GPW * 16 * EE; i += 256) {
      int c = i % EE, row = i / EE, g = row >> 4, l = row & 15, h = c >> 6;
      const float* a = sScr + ((g * HH + h) * 16 + l) * 16;
      float acc = 0.f;
      #pragma unroll
      for (int m = 0; m < 16; ++m) acc += a[m] * bf2f(sV[(g * 16 + m) * EE + c]);
      sSeq[row * EE + c] = f2bf(acc);
    }
    __syncthreads();
    // out projection: aL = o @ o_w.T + o_b
    gemm_sw<2>(rowSeq, Ow[L], ob[L], aDst[L], EE, tid);
    __syncthreads();
  }

  // ---------- final 1-head attention (only token 0 output needed) ----------
  auto rowAtt = [&](int r) -> const unsigned short* {
    int g = r >> 5, t = r & 31;
    const unsigned short* base = (t < 16) ? sAin : sAout;
    return base + (g * 16 + (t & 15)) * EE;
  };
  gemm_sw<4>(rowAtt, SW + 8 * MSZ, fin_qkv_b + EE,     sK2f, EE, tid);  // k2f
  gemm_sw<4>(rowAtt, SW + 9 * MSZ, fin_qkv_b + 2 * EE, sV2f, EE, tid);  // v2f
  // q0 (one row per node) directly from fp32 weights
  for (int i = tid; i < GPW * EE; i += 256) {
    int g = i / EE, c = i % EE;
    const unsigned short* a0 = sAin + (g * 16) * EE;
    const AS1 float* wrow = (const AS1 float*)(fin_qkv_w + c * EE);    // wq row c
    float acc = ((const AS1 float*)fin_qkv_b)[c];
    #pragma unroll 8
    for (int j = 0; j < EE; ++j) acc += bf2f(a0[j]) * wrow[j];
    sScr[i] = acc;
  }
  __syncthreads();
  if (tid < GPW * 32) {                                // scores over 32 tokens
    int g = tid >> 5, m = tid & 31;
    const float* q0 = sScr + g * EE;
    const unsigned short* kr = sK2f + (g * 32 + m) * EE;
    float acc = 0.f;
    #pragma unroll 8
    for (int c = 0; c < EE; ++c) acc += q0[c] * bf2f(kr[c]);
    sScr[1280 + tid] = acc * rsqrtf((float)EE);
  }
  __syncthreads();
  if (tid < GPW) {                                     // softmax(32)
    float* row = sScr + 1280 + tid * 32;
    float mx = row[0];
    for (int m = 1; m < 32; ++m) mx = fmaxf(mx, row[m]);
    float s = 0.f;
    for (int m = 0; m < 32; ++m) { float e = __expf(row[m] - mx); row[m] = e; s += e; }
    float inv = 1.f / s;
    for (int m = 0; m < 32; ++m) row[m] *= inv;
  }
  __syncthreads();
  for (int i = tid; i < GPW * EE; i += 256) {          // o0 = a @ v2f
    int g = i / EE, c = i % EE;
    const float* a = sScr + 1280 + g * 32;
    float acc = 0.f;
    #pragma unroll
    for (int m = 0; m < 32; ++m) acc += a[m] * bf2f(sV2f[(g * 32 + m) * EE + c]);
    sScr[640 + i] = acc;
  }
  __syncthreads();
  for (int i = tid; i < GPW * EE; i += 256) {          // fin0 = o0 @ o_w.T + o_b
    int g = i / EE, c = i % EE;
    const float* o0 = sScr + 640 + g * EE;
    const AS1 float* wrow = (const AS1 float*)(fin_o_w + c * EE);
    float acc = ((const AS1 float*)fin_o_b)[c];
    #pragma unroll 8
    for (int k = 0; k < EE; ++k) acc += o0[k] * wrow[k];
    sScr[1344 + i] = acc;
  }
  __syncthreads();
  if (tid < GPW * OUTD) {                              // elu(fin0 @ W)
    int g = tid / OUTD, j = tid % OUTD;
    const float* f0 = sScr + 1344 + g * EE;
    const AS1 float* Wg = (const AS1 float*)Wfin;
    float acc = 0.f;
    #pragma unroll 8
    for (int n = 0; n < EE; ++n) acc += f0[n] * Wg[n * OUTD + j];
    ((AS1 float*)out)[(node0 + g) * OUTD + j] = (acc > 0.f) ? acc : (__expf(acc) - 1.f);
  }
}

// ---------------- launcher ----------------
extern "C" void kernel_launch(void* const* d_in, const int* in_sizes, int n_in,
                              void* d_out, int out_size, void* d_ws, size_t ws_size,
                              hipStream_t stream) {
  const float* X        = (const float*)d_in[0];
  const int*   in_idx   = (const int*)d_in[1];
  const int*   out_idx  = (const int*)d_in[2];
  const float* in_Wq    = (const float*)d_in[3];
  const float* in_Wk    = (const float*)d_in[4];
  const float* in_Wv    = (const float*)d_in[5];
  const float* in_qkv_w = (const float*)d_in[6];
  const float* in_qkv_b = (const float*)d_in[7];
  const float* in_o_w   = (const float*)d_in[8];
  const float* in_o_b   = (const float*)d_in[9];
  const float* out_Wq   = (const float*)d_in[10];
  const float* out_Wk   = (const float*)d_in[11];
  const float* out_Wv   = (const float*)d_in[12];
  const float* out_qkv_w= (const float*)d_in[13];
  const float* out_qkv_b= (const float*)d_in[14];
  const float* out_o_w  = (const float*)d_in[15];
  const float* out_o_b  = (const float*)d_in[16];
  const float* fin_qkv_w= (const float*)d_in[17];
  const float* fin_qkv_b= (const float*)d_in[18];
  const float* fin_o_w  = (const float*)d_in[19];
  const float* fin_o_b  = (const float*)d_in[20];
  const float* Wfin     = (const float*)d_in[21];

  unsigned short* Xbf = (unsigned short*)d_ws;
  unsigned short* SW  = Xbf + (size_t)NN * EE;

  k_xbf<<<(NN * EE + 255) / 256, 256, 0, stream>>>(X, Xbf, NN * EE);
  k_combined<<<dim3((EE * EE + 255) / 256, 6), 256, 0, stream>>>(
      in_Wq, in_Wk, in_Wv, in_qkv_w, out_Wq, out_Wk, out_Wv, out_qkv_w, SW);
  k_transpose_sw<<<dim3((EE * EE + 255) / 256, 4), 256, 0, stream>>>(
      in_o_w, out_o_w, fin_qkv_w, SW);
  k_main<<<NN / GPW, 256, SMEM_BYTES, stream>>>(
      Xbf, SW, in_idx, out_idx, in_qkv_b, in_o_b, out_qkv_b, out_o_b,
      fin_qkv_w, fin_qkv_b, fin_o_w, fin_o_b, Wfin, (float*)d_out);
}
